// TransformerDecoderBlock_17428977287761
// MI455X (gfx1250) — compile-verified, matrix operand order checked by reading, not run on hardware
//
#include <hip/hip_runtime.h>
#include <cstdint>
#include <cstddef>

// ---------------------------------------------------------------------------
// CDNA5 (gfx1250) wave32 WMMA transformer decoder block.
// Heavy math: v_wmma_f32_16x16x32_bf16 (f32 accumulate).
// A-operand tiles staged into LDS via the Tensor Data Mover (tensor_load_to_lds,
// TENSORcnt-tracked, double buffered). Weights via clause-grouped b128 loads.
// ---------------------------------------------------------------------------

typedef __attribute__((ext_vector_type(16))) __bf16 v16bf;
typedef __attribute__((ext_vector_type(8)))  float  v8f;
typedef __attribute__((ext_vector_type(4)))  unsigned int v4u;
typedef __attribute__((ext_vector_type(8)))  int    v8i;
typedef __attribute__((ext_vector_type(4)))  int    v4i;

static constexpr int   Bb    = 2;
static constexpr int   Ss    = 2048;
static constexpr int   Dd    = 1024;
static constexpr int   Hh    = 16;
static constexpr int   HDh   = 64;
static constexpr int   DFFc  = 4096;
static constexpr int   RANKc = 32;
static constexpr int   Mrows = Bb * Ss;          // 4096
static constexpr float LSCALE = 1.0f / RANKc;
static constexpr float EPSc   = 1e-6f;
static constexpr float NEGc   = -1e9f;

// LDS tile: 32 rows x 32 bf16, padded to 40 bf16 (80B) row stride so the
// 16B fragment reads are spread across banks. Matches TDM pad_interval=16
// DWORDs (one 64B row) + pad_amount=4 DWORDs (16B).
static constexpr int TILE_LDS_STRIDE = 40;

#if __has_builtin(__builtin_amdgcn_tensor_load_to_lds)
#define HAVE_TDM 1
#else
#define HAVE_TDM 0
#endif

#define DEVFN __device__ __forceinline__

// ---- WMMA fragment helpers -------------------------------------------------
// 16-bit A/B fragment layout (ISA 7.12.2): lane L holds row (L&15); element j
// of the 16-element fragment is K = klo + j + (j>=8 ? 8 : 0), klo = (L>=16)*8.

DEVFN v16bf load_frag_bf(const __bf16* rowp, int klo) {
  union { uint4 q[2]; v16bf v; } u;
  u.q[0] = *(const uint4*)(rowp + klo);
  u.q[1] = *(const uint4*)(rowp + klo + 16);
  return u.v;
}

DEVFN v16bf load_frag_f32(const float* rowp, int klo) {
  v16bf v;
#pragma unroll
  for (int j = 0; j < 16; ++j) {
    int K = klo + j + ((j >= 8) ? 8 : 0);
    v[j] = (__bf16)rowp[K];
  }
  return v;
}

DEVFN v16bf load_frag_strided(const __bf16* base, int ld, int col, int klo) {
  v16bf v;
#pragma unroll
  for (int j = 0; j < 16; ++j) {
    int K = klo + j + ((j >= 8) ? 8 : 0);
    v[j] = base[(size_t)K * ld + col];
  }
  return v;
}

DEVFN v8f wmma_bf16(v16bf a, v16bf b, v8f c) {
  // 8-arg form: (neg_a, A, neg_b, B, c_mod, C, reuse_a, reuse_b)
  return __builtin_amdgcn_wmma_f32_16x16x32_bf16(false, a, false, b,
                                                 (short)0, c, false, false);
}

#if HAVE_TDM
// ---- Tensor Data Mover: DMA a 32x32 bf16 tile (row stride Kdim elems) into
// LDS at byte offset lds_off, padded to an 80B LDS row stride. D# per ISA 8.x.
// amdgpu-toolchain (clang-23) 6-arg form:
//   (uint32x4 g0, int32x8 g1, int32x4 g2, int32x4 g3, int32x8 g4, i32 cpol)
DEVFN void tdm_load_tile32x32(unsigned lds_off, const __bf16* gptr, int Kdim) {
  const unsigned long long ga = (unsigned long long)(size_t)gptr;
  v4u g0;
  g0[0] = 1u;                                   // count=1, user descriptor
  g0[1] = lds_off;                              // lds_addr (bytes)
  g0[2] = (unsigned)(ga & 0xffffffffu);         // global_addr[31:0]
  g0[3] = (unsigned)((ga >> 32) & 0x1ffffffu)   // global_addr[56:32]
          | (2u << 30);                         // type=2 ("image")
  const unsigned td0 = 32u, td1 = 32u;          // tensor dims (>= tile dims)
  const unsigned long long st0 = (unsigned long long)Kdim;  // dim0 stride
  v8i g1;
  g1[0] = (int)((1u << 16)      // data_size = 1 -> 2 bytes
                | (1u << 20)    // pad_enable
                | (3u << 22)    // pad_interval: 16 DWORDs (one 64B row)
                | (3u << 25));  // pad_amount: 4 DWORDs (16B)
  g1[1] = (int)((td0 & 0xffffu) << 16);                       // tensor_dim0 lo
  g1[2] = (int)(((td0 >> 16) & 0xffffu) | ((td1 & 0xffffu) << 16));
  g1[3] = (int)(((td1 >> 16) & 0xffffu) | (32u << 16));       // tile_dim0 = 32
  g1[4] = (int)(32u);                                         // tile_dim1 = 32
  g1[5] = (int)(st0 & 0xffffffffu);                           // stride0 lo
  g1[6] = (int)((st0 >> 32) & 0xffffu);                       // stride0 hi
  g1[7] = 0;
  v4i gz4 = {0, 0, 0, 0};                        // groups 2/3 unused (2D)
  v8i gz8 = {0, 0, 0, 0, 0, 0, 0, 0};            // trailing group unused
  __builtin_amdgcn_tensor_load_to_lds(g0, g1, gz4, gz4, gz8, 0);
}
#endif

// ---- fp32 -> bf16 convert (weights, once per launch) -----------------------
__global__ void f32_to_bf16_kernel(const float* __restrict__ in,
                                   __bf16* __restrict__ out, int n) {
  for (int i = blockIdx.x * blockDim.x + threadIdx.x; i < n;
       i += gridDim.x * blockDim.x)
    out[i] = (__bf16)in[i];
}

// ---- LayerNorm -> bf16 -----------------------------------------------------
__global__ void ln_bf16_kernel(const float* __restrict__ x,
                               const float* __restrict__ g,
                               const float* __restrict__ b,
                               __bf16* __restrict__ out) {
  const int row = blockIdx.x;
  const int t   = threadIdx.x;            // 256 threads, 4 elems each
  const float* xr = x + (size_t)row * Dd;
  float4 v = ((const float4*)xr)[t];
  __shared__ float sh[256];
  sh[t] = v.x + v.y + v.z + v.w;
  __syncthreads();
  for (int off = 128; off > 0; off >>= 1) {
    if (t < off) sh[t] += sh[t + off];
    __syncthreads();
  }
  float mu = sh[0] * (1.0f / Dd);
  __syncthreads();
  float dx = v.x - mu, dy = v.y - mu, dz = v.z - mu, dw = v.w - mu;
  sh[t] = dx * dx + dy * dy + dz * dz + dw * dw;
  __syncthreads();
  for (int off = 128; off > 0; off >>= 1) {
    if (t < off) sh[t] += sh[t + off];
    __syncthreads();
  }
  float rstd = rsqrtf(sh[0] * (1.0f / Dd) + EPSc);
  __bf16* o = out + (size_t)row * Dd + t * 4;
  const float* gp = g + t * 4;
  const float* bp = b + t * 4;
  o[0] = (__bf16)(dx * rstd * gp[0] + bp[0]);
  o[1] = (__bf16)(dy * rstd * gp[1] + bp[1]);
  o[2] = (__bf16)(dz * rstd * gp[2] + bp[2]);
  o[3] = (__bf16)(dw * rstd * gp[3] + bp[3]);
}

// ---- LoRA down-projection: T = (X @ A) * (1/rank), T is (M x 32) f32 -------
__global__ void lora_t_kernel(const __bf16* __restrict__ X,
                              const float* __restrict__ A,
                              float* __restrict__ T, int K) {
  const int row = blockIdx.x;
  const int n   = threadIdx.x;            // 32 lanes = 32 rank cols
  const __bf16* xr = X + (size_t)row * K;
  float s = 0.f;
  for (int k = 0; k < K; k += 4) {
    s += (float)xr[k + 0] * A[(size_t)(k + 0) * RANKc + n];
    s += (float)xr[k + 1] * A[(size_t)(k + 1) * RANKc + n];
    s += (float)xr[k + 2] * A[(size_t)(k + 2) * RANKc + n];
    s += (float)xr[k + 3] * A[(size_t)(k + 3) * RANKc + n];
  }
  T[(size_t)row * RANKc + n] = s * LSCALE;
}

// ---- Generic WMMA GEMM: C = X(bf16) @ W(bf16,(N,K) rowmajor)^T + LoRA ------
// MODE 0: out bf16 (QKV)
// MODE 1: outF = add + lsv*C  (attention out-proj residual -> x2)
// MODE 2: dual half (gate/act), outB = silu(g)*a  (SwiGLU -> u)
// MODE 3: outF = add + lsv*C  (FFN down residual -> final output)
template <int MODE>
__global__ void gemm_wmma_kernel(const __bf16* __restrict__ X,
                                 const __bf16* __restrict__ W,
                                 const float*  __restrict__ T,
                                 const __bf16* __restrict__ Bl,
                                 const float*  __restrict__ addv,
                                 const float*  __restrict__ lsv,
                                 float*        __restrict__ outF,
                                 __bf16*       __restrict__ outB,
                                 int Kdim, int Ndim, int ldbl) {
  const int lane = threadIdx.x & 31;
  const int half = lane >> 4;
  const int l16  = lane & 15;
  const int klo  = half * 8;
  const int m0   = blockIdx.x * 32;
  const int n0   = blockIdx.y * 32;

  const __bf16* wr0 = W + (size_t)(n0 + l16) * Kdim;
  const __bf16* wr1 = wr0 + (size_t)16 * Kdim;
  const __bf16* wa0 = wr0 + (size_t)DFFc * Kdim;   // act half (MODE 2)
  const __bf16* wa1 = wa0 + (size_t)16 * Kdim;

  v8f zero = {0.f, 0.f, 0.f, 0.f, 0.f, 0.f, 0.f, 0.f};
  v8f acc[2][2]  = {{zero, zero}, {zero, zero}};
  v8f acc2[2][2] = {{zero, zero}, {zero, zero}};

#if HAVE_TDM
  // Double-buffered TDM staging of the A (activation) tile into LDS.
  __shared__ __align__(128) __bf16 tileA[2][32 * TILE_LDS_STRIDE];
  const unsigned lds_base[2] = {
      (unsigned)(size_t)&tileA[0][0],
      (unsigned)(size_t)&tileA[1][0]};
  const __bf16* xg = X + (size_t)m0 * Kdim;
  tdm_load_tile32x32(lds_base[0], xg, Kdim);
#else
  const __bf16* xr0 = X + (size_t)(m0 + l16) * Kdim;
  const __bf16* xr1 = xr0 + (size_t)16 * Kdim;
#endif

  for (int k = 0; k < Kdim; k += 32) {
    __builtin_prefetch(wr0 + k + 128, 0, 1);
    __builtin_prefetch(wr1 + k + 128, 0, 1);
#if HAVE_TDM
    const int cur = (k >> 5) & 1;
    const bool more = (k + 32) < Kdim;
    if (more) {
      // Fence: prior LDS reads from the other buffer must land before the
      // TDM overwrites it; then kick off the next tile DMA.
      asm volatile("s_wait_dscnt 0" ::: "memory");
      tdm_load_tile32x32(lds_base[1 - cur], xg + (k + 32), Kdim);
      __builtin_amdgcn_s_wait_tensorcnt(1);   // oldest (current) tile done
    } else {
      __builtin_amdgcn_s_wait_tensorcnt(0);
    }
    v16bf a0 = load_frag_bf(&tileA[cur][(l16)      * TILE_LDS_STRIDE], klo);
    v16bf a1 = load_frag_bf(&tileA[cur][(16 + l16) * TILE_LDS_STRIDE], klo);
#else
    v16bf a0 = load_frag_bf(xr0 + k, klo);
    v16bf a1 = load_frag_bf(xr1 + k, klo);
#endif
    v16bf b0 = load_frag_bf(wr0 + k, klo);
    v16bf b1 = load_frag_bf(wr1 + k, klo);
    acc[0][0] = wmma_bf16(a0, b0, acc[0][0]);
    acc[0][1] = wmma_bf16(a0, b1, acc[0][1]);
    acc[1][0] = wmma_bf16(a1, b0, acc[1][0]);
    acc[1][1] = wmma_bf16(a1, b1, acc[1][1]);
    if (MODE == 2) {
      v16bf c0 = load_frag_bf(wa0 + k, klo);
      v16bf c1 = load_frag_bf(wa1 + k, klo);
      acc2[0][0] = wmma_bf16(a0, c0, acc2[0][0]);
      acc2[0][1] = wmma_bf16(a0, c1, acc2[0][1]);
      acc2[1][0] = wmma_bf16(a1, c0, acc2[1][0]);
      acc2[1][1] = wmma_bf16(a1, c1, acc2[1][1]);
    }
  }

  // LoRA epilogue: one rank-32 WMMA k-step, accumulated into the same tiles.
  {
    v16bf ta0 = load_frag_f32(T + (size_t)(m0 + l16) * RANKc, klo);
    v16bf ta1 = load_frag_f32(T + (size_t)(m0 + 16 + l16) * RANKc, klo);
    v16bf lb0 = load_frag_strided(Bl, ldbl, n0 + l16, klo);
    v16bf lb1 = load_frag_strided(Bl, ldbl, n0 + 16 + l16, klo);
    acc[0][0] = wmma_bf16(ta0, lb0, acc[0][0]);
    acc[0][1] = wmma_bf16(ta0, lb1, acc[0][1]);
    acc[1][0] = wmma_bf16(ta1, lb0, acc[1][0]);
    acc[1][1] = wmma_bf16(ta1, lb1, acc[1][1]);
    if (MODE == 2) {
      v16bf la0 = load_frag_strided(Bl, ldbl, DFFc + n0 + l16, klo);
      v16bf la1 = load_frag_strided(Bl, ldbl, DFFc + n0 + 16 + l16, klo);
      acc2[0][0] = wmma_bf16(ta0, la0, acc2[0][0]);
      acc2[0][1] = wmma_bf16(ta0, la1, acc2[0][1]);
      acc2[1][0] = wmma_bf16(ta1, la0, acc2[1][0]);
      acc2[1][1] = wmma_bf16(ta1, la1, acc2[1][1]);
    }
  }

  // Store: C/D layout => row = base + r + 8*half, col = base + l16.
#pragma unroll
  for (int mi = 0; mi < 2; ++mi)
#pragma unroll
    for (int ni = 0; ni < 2; ++ni)
#pragma unroll
      for (int r = 0; r < 8; ++r) {
        const int row = m0 + mi * 16 + r + 8 * half;
        const int col = n0 + ni * 16 + l16;
        const size_t idx = (size_t)row * Ndim + col;
        const float val = acc[mi][ni][r];
        if (MODE == 0) {
          outB[idx] = (__bf16)val;
        } else if (MODE == 1 || MODE == 3) {
          outF[idx] = addv[idx] + lsv[col] * val;
        } else {  // MODE == 2: SwiGLU
          const float gte = val;
          const float act = acc2[mi][ni][r];
          const float sig = 1.0f / (1.0f + __expf(-gte));
          outB[idx] = (__bf16)(gte * sig * act);
        }
      }
}

// ---- Flash attention with analytic ALiBi + causal mask ---------------------
// One wave per (b, h, 16-row q tile). qkv is bf16 (Mrows x 3072),
// col layout = c*1024 + h*64 + d. Output: attn bf16 (Mrows x 1024).
__global__ void attn_flash_kernel(const __bf16* __restrict__ qkv,
                                  __bf16* __restrict__ attn) {
  const int lane = threadIdx.x & 31;
  const int half = lane >> 4;
  const int l16  = lane & 15;
  const int klo  = half * 8;
  const int bh   = blockIdx.x;
  const int b    = bh / Hh;
  const int h    = bh % Hh;
  const int q0   = blockIdx.y * 16;
  const float slope = exp2f(-0.5f * (float)(h + 1));
  const size_t rs = 3 * Dd;

  const __bf16* base  = qkv + (size_t)b * Ss * rs;
  const __bf16* qrow  = base + (size_t)(q0 + l16) * rs + h * HDh;
  const __bf16* kbase = base + Dd + h * HDh;
  const __bf16* vbase = base + 2 * Dd + h * HDh;

  const v16bf qf0 = load_frag_bf(qrow, klo);
  const v16bf qf1 = load_frag_bf(qrow + 32, klo);

  v8f zero = {0.f, 0.f, 0.f, 0.f, 0.f, 0.f, 0.f, 0.f};
  v8f o[4] = {zero, zero, zero, zero};
  float rowm[8], rowl[8];
#pragma unroll
  for (int r = 0; r < 8; ++r) { rowm[r] = -__builtin_inff(); rowl[r] = 0.f; }

  __shared__ float plds[16 * 32];

  const int kend = q0 + 16;                 // causal: keys <= last q row
  for (int kt = 0; kt < kend; kt += 32) {
    float sc[2][8];
#pragma unroll
    for (int nt = 0; nt < 2; ++nt) {
      const int scol = kt + nt * 16 + l16;  // this lane's key index
      const int scl  = scol < Ss ? scol : Ss - 1;
      const __bf16* krow = kbase + (size_t)scl * rs;
      v16bf kf0 = load_frag_bf(krow, klo);
      v16bf kf1 = load_frag_bf(krow + 32, klo);
      v8f sa = zero;
      sa = wmma_bf16(qf0, kf0, sa);
      sa = wmma_bf16(qf1, kf1, sa);
#pragma unroll
      for (int r = 0; r < 8; ++r) {
        const int qi = q0 + r + 8 * half;
        const float bias =
            (scol <= qi) ? slope * (float)(scol - qi) : NEGc;
        sc[nt][r] = sa[r] * 0.125f + bias;   // 1/sqrt(64)
      }
    }

    // online softmax update (row stats live per (r, half); reduce 16 lanes)
#pragma unroll
    for (int r = 0; r < 8; ++r) {
      float mt = fmaxf(sc[0][r], sc[1][r]);
      for (int msk = 1; msk < 16; msk <<= 1)
        mt = fmaxf(mt, __shfl_xor(mt, msk, 32));
      const float mnew = fmaxf(rowm[r], mt);
      const float corr = __expf(rowm[r] - mnew);
      rowm[r] = mnew;
      const float p0 = __expf(sc[0][r] - mnew);
      const float p1 = __expf(sc[1][r] - mnew);
      float rsum = p0 + p1;
      for (int msk = 1; msk < 16; msk <<= 1)
        rsum += __shfl_xor(rsum, msk, 32);
      rowl[r] = rowl[r] * corr + rsum;
#pragma unroll
      for (int dt = 0; dt < 4; ++dt) o[dt][r] *= corr;
      plds[(r + 8 * half) * 32 + l16]      = p0;
      plds[(r + 8 * half) * 32 + 16 + l16] = p1;
    }
    __syncthreads();

    // P (C-layout) -> A-fragment via LDS transpose
    v16bf pf;
#pragma unroll
    for (int j = 0; j < 16; ++j) {
      const int K = klo + j + ((j >= 8) ? 8 : 0);
      pf[j] = (__bf16)plds[l16 * 32 + K];
    }

    // O += P @ V  (V B-fragments gathered: lane = output dim d)
#pragma unroll
    for (int dt = 0; dt < 4; ++dt) {
      v16bf vf;
#pragma unroll
      for (int j = 0; j < 16; ++j) {
        const int K = klo + j + ((j >= 8) ? 8 : 0);
        int srow = kt + K;
        if (srow > Ss - 1) srow = Ss - 1;
        vf[j] = vbase[(size_t)srow * rs + dt * 16 + l16];
      }
      o[dt] = wmma_bf16(pf, vf, o[dt]);
    }
    __syncthreads();
  }

#pragma unroll
  for (int dt = 0; dt < 4; ++dt)
#pragma unroll
    for (int r = 0; r < 8; ++r) {
      const int row = q0 + r + 8 * half;
      const float val = o[dt][r] / rowl[r];
      attn[(size_t)(b * Ss + row) * Dd + h * HDh + dt * 16 + l16] =
          (__bf16)val;
    }
}

// ---------------------------------------------------------------------------
extern "C" void kernel_launch(void* const* d_in, const int* in_sizes, int n_in,
                              void* d_out, int out_size, void* d_ws,
                              size_t ws_size, hipStream_t stream) {
  (void)in_sizes; (void)n_in; (void)out_size; (void)ws_size;

  const float* x    = (const float*)d_in[0];
  // d_in[1] = alibi_combined_bias (268 MB): recomputed analytically, unused.
  const float* ln1g = (const float*)d_in[2];
  const float* ln1b = (const float*)d_in[3];
  const float* Wqkv = (const float*)d_in[4];
  const float* Aqkv = (const float*)d_in[5];
  const float* Bqkv = (const float*)d_in[6];
  const float* Wo   = (const float*)d_in[7];
  const float* Ao   = (const float*)d_in[8];
  const float* Bo   = (const float*)d_in[9];
  const float* ln2g = (const float*)d_in[10];
  const float* ln2b = (const float*)d_in[11];
  const float* W1   = (const float*)d_in[12];
  const float* A1   = (const float*)d_in[13];
  const float* B1   = (const float*)d_in[14];
  const float* W2   = (const float*)d_in[15];
  const float* A2   = (const float*)d_in[16];
  const float* B2   = (const float*)d_in[17];
  const float* ls1  = (const float*)d_in[18];
  const float* ls2  = (const float*)d_in[19];
  float* out = (float*)d_out;

  // workspace carve-up (256B aligned)
  char* wp = (char*)d_ws;
  auto alloc = [&](size_t bytes) -> char* {
    char* p = wp;
    wp += (bytes + 255) & ~(size_t)255;
    return p;
  };
  __bf16* wqkv_bf = (__bf16*)alloc((size_t)3 * Dd * Dd * 2);
  __bf16* wo_bf   = (__bf16*)alloc((size_t)Dd * Dd * 2);
  __bf16* w1_bf   = (__bf16*)alloc((size_t)2 * DFFc * Dd * 2);
  __bf16* w2_bf   = (__bf16*)alloc((size_t)Dd * DFFc * 2);
  __bf16* bqkv_bf = (__bf16*)alloc((size_t)RANKc * 3 * Dd * 2);
  __bf16* bo_bf   = (__bf16*)alloc((size_t)RANKc * Dd * 2);
  __bf16* b1_bf   = (__bf16*)alloc((size_t)RANKc * 2 * DFFc * 2);
  __bf16* b2_bf   = (__bf16*)alloc((size_t)RANKc * Dd * 2);
  __bf16* xn1_bf  = (__bf16*)alloc((size_t)Mrows * Dd * 2);
  __bf16* qkv_bf  = (__bf16*)alloc((size_t)Mrows * 3 * Dd * 2);
  __bf16* attn_bf = (__bf16*)alloc((size_t)Mrows * Dd * 2);
  float*  x2      = (float*) alloc((size_t)Mrows * Dd * 4);
  __bf16* xn2_bf  = (__bf16*)alloc((size_t)Mrows * Dd * 2);
  __bf16* u_bf    = (__bf16*)alloc((size_t)Mrows * DFFc * 2);
  float*  Tq      = (float*) alloc((size_t)Mrows * RANKc * 4);
  float*  To      = (float*) alloc((size_t)Mrows * RANKc * 4);
  float*  T1      = (float*) alloc((size_t)Mrows * RANKc * 4);
  float*  T2      = (float*) alloc((size_t)Mrows * RANKc * 4);

  const dim3 cvB(256), cvG(2048);
  f32_to_bf16_kernel<<<cvG, cvB, 0, stream>>>(Wqkv, wqkv_bf, 3 * Dd * Dd);
  f32_to_bf16_kernel<<<cvG, cvB, 0, stream>>>(Wo,   wo_bf,   Dd * Dd);
  f32_to_bf16_kernel<<<cvG, cvB, 0, stream>>>(W1,   w1_bf,   2 * DFFc * Dd);
  f32_to_bf16_kernel<<<cvG, cvB, 0, stream>>>(W2,   w2_bf,   Dd * DFFc);
  f32_to_bf16_kernel<<<cvG, cvB, 0, stream>>>(Bqkv, bqkv_bf, RANKc * 3 * Dd);
  f32_to_bf16_kernel<<<cvG, cvB, 0, stream>>>(Bo,   bo_bf,   RANKc * Dd);
  f32_to_bf16_kernel<<<cvG, cvB, 0, stream>>>(B1,   b1_bf,   RANKc * 2 * DFFc);
  f32_to_bf16_kernel<<<cvG, cvB, 0, stream>>>(B2,   b2_bf,   RANKc * Dd);

  // 1) LN1
  ln_bf16_kernel<<<dim3(Mrows), dim3(256), 0, stream>>>(x, ln1g, ln1b, xn1_bf);
  // 2) QKV = xn1 @ Wqkv^T + lora
  lora_t_kernel<<<dim3(Mrows), dim3(32), 0, stream>>>(xn1_bf, Aqkv, Tq, Dd);
  gemm_wmma_kernel<0><<<dim3(Mrows / 32, 3 * Dd / 32), dim3(32), 0, stream>>>(
      xn1_bf, wqkv_bf, Tq, bqkv_bf, nullptr, nullptr, nullptr, qkv_bf,
      Dd, 3 * Dd, 3 * Dd);
  // 3) flash attention (analytic ALiBi + causal)
  attn_flash_kernel<<<dim3(Bb * Hh, Ss / 16), dim3(32), 0, stream>>>(qkv_bf,
                                                                     attn_bf);
  // 4) x2 = x + ls1 * (attn @ Wo^T + lora)
  lora_t_kernel<<<dim3(Mrows), dim3(32), 0, stream>>>(attn_bf, Ao, To, Dd);
  gemm_wmma_kernel<1><<<dim3(Mrows / 32, Dd / 32), dim3(32), 0, stream>>>(
      attn_bf, wo_bf, To, bo_bf, x, ls1, x2, nullptr, Dd, Dd, Dd);
  // 5) LN2
  ln_bf16_kernel<<<dim3(Mrows), dim3(256), 0, stream>>>(x2, ln2g, ln2b, xn2_bf);
  // 6) u = silu(gate) * act,  [gate|act] = xn2 @ W1^T + lora
  lora_t_kernel<<<dim3(Mrows), dim3(32), 0, stream>>>(xn2_bf, A1, T1, Dd);
  gemm_wmma_kernel<2><<<dim3(Mrows / 32, DFFc / 32), dim3(32), 0, stream>>>(
      xn2_bf, w1_bf, T1, b1_bf, nullptr, nullptr, nullptr, u_bf,
      Dd, DFFc, 2 * DFFc);
  // 7) out = x2 + ls2 * (u @ W2^T + lora)
  lora_t_kernel<<<dim3(Mrows), dim3(32), 0, stream>>>(u_bf, A2, T2, DFFc);
  gemm_wmma_kernel<3><<<dim3(Mrows / 32, Dd / 32), dim3(32), 0, stream>>>(
      u_bf, w2_bf, T2, b2_bf, x2, ls2, out, nullptr, DFFc, Dd, Dd);
}
